// GCN_20847771254960
// MI455X (gfx1250) — matrix operand order repaired and checked by reading, not compiled, not run on hardware
//
#include <hip/hip_runtime.h>

typedef __attribute__((ext_vector_type(16))) _Float16 v16h;
typedef __attribute__((ext_vector_type(8)))  float    v8f;
typedef __attribute__((ext_vector_type(4)))  float    v4f;

#define NN 50000
#define EE 800000
#define CC 128
#define LL 4

// ---------------- preprocessing kernels ----------------

// Wt[l][n][k] = (f16) W[l][k][n]   (column-major transpose for contiguous B-tile loads)
__global__ void gcn_wcvt(const float* __restrict__ W, _Float16* __restrict__ Wt, int total) {
    int i = blockIdx.x * blockDim.x + threadIdx.x;
    if (i >= total) return;
    int l   = i >> 14;            // / (128*128)
    int rem = i & 16383;
    int n   = rem >> 7;
    int k   = rem & 127;
    Wt[i] = (_Float16)W[l * 16384 + k * 128 + n];
}

__global__ void gcn_deg_init(float* __restrict__ deg, int n) {
    int i = blockIdx.x * blockDim.x + threadIdx.x;
    if (i < n) deg[i] = 1.0f;                       // self-loop contribution
}

__global__ void gcn_deg_count(const int* __restrict__ cols, float* __restrict__ deg, int e) {
    int i = blockIdx.x * blockDim.x + threadIdx.x;
    if (i < e) atomicAdd(&deg[cols[i]], 1.0f);
}

__global__ void gcn_deg_rsqrt(float* __restrict__ deg, int n) {
    int i = blockIdx.x * blockDim.x + threadIdx.x;
    if (i < n) deg[i] = rsqrtf(deg[i]);             // deg >= 1 always (self-loops)
}

__global__ void gcn_norm(const int* __restrict__ rows, const int* __restrict__ cols,
                         const float* __restrict__ dinv, float* __restrict__ nrm, int e) {
    int i = blockIdx.x * blockDim.x + threadIdx.x;
    if (i < e) nrm[i] = dinv[rows[i]] * dinv[cols[i]];
}

// ---------------- WMMA GEMM: H[N,128] = X[N,128] @ W[128,128] ----------------
// Whole W (32KB f16, col-major) staged in LDS once per workgroup; one wave per
// 16-row strip; 8 column tiles of 16; K swept in 4 blocks of 32.
__global__ void __launch_bounds__(256)
gcn_gemm_wmma(const float* __restrict__ X, const _Float16* __restrict__ Wt,
              float* __restrict__ H, int nStrips) {
    __shared__ _Float16 Bs[CC * CC];                // 32 KB of 320 KB WGP LDS

    // cooperative fill: 2048 x 16B = 32KB, 8 iterations per thread
    {
        const v4f* src = (const v4f*)Wt;
        v4f* dst = (v4f*)Bs;
#pragma unroll
        for (int i = threadIdx.x; i < (CC * CC) / 8; i += 256)
            dst[i] = src[i];
    }
    __syncthreads();

    int wave = (blockIdx.x * blockDim.x + threadIdx.x) >> 5;
    int lane = threadIdx.x & 31;
    if (wave >= nStrips) return;                    // wave-uniform: EXEC all-ones at WMMA

    const int sub = lane >> 4;                      // 0 or 1 (lane half)
    const int l16 = lane & 15;
    const int m0  = wave * 16;
    const int m   = m0 + l16;                       // A row held by this lane
    const float* arow = X + (size_t)m * CC;

    v8f acc[8] = {};

#pragma unroll
    for (int kb = 0; kb < 4; ++kb) {
        const int k0 = kb * 32;
        // A tile 16x32 f16: halves 0..15 -> K = k0 + sub*16 + i
        v4f a0 = *(const v4f*)(arow + k0 + sub * 16);
        v4f a1 = *(const v4f*)(arow + k0 + sub * 16 + 4);
        v4f a2 = *(const v4f*)(arow + k0 + sub * 16 + 8);
        v4f a3 = *(const v4f*)(arow + k0 + sub * 16 + 12);
        v16h a;
#pragma unroll
        for (int i = 0; i < 4; ++i) {
            a[i]      = (_Float16)a0[i];
            a[4 + i]  = (_Float16)a1[i];
            a[8 + i]  = (_Float16)a2[i];
            a[12 + i] = (_Float16)a3[i];
        }
#pragma unroll
        for (int nt = 0; nt < 8; ++nt) {
            // B tile 32x16 f16 from LDS: lane = column n = l16,
            // K = k0 + sub*16 + {0..15}, contiguous in col-major Bs.
            const _Float16* bcol = Bs + (nt * 16 + l16) * CC + k0 + sub * 16;
            v16h b = *(const v16h*)bcol;
            acc[nt] = __builtin_amdgcn_wmma_f32_16x16x32_f16(
                false, a, false, b, (short)0, acc[nt], false, false);
        }
    }

    // C/D layout: VGPR r -> M = m0 + r + sub*8, N = nt*16 + l16
#pragma unroll
    for (int nt = 0; nt < 8; ++nt) {
#pragma unroll
        for (int r = 0; r < 8; ++r) {
            int M = m0 + r + sub * 8;
            int Nc = nt * 16 + l16;
            H[(size_t)M * CC + Nc] = acc[nt][r];
        }
    }
}

// ---------------- scatter-add: one wave per edge, 4 channels per lane ----------------
__global__ void gcn_scatter(const float* __restrict__ H, const int* __restrict__ rows,
                            const int* __restrict__ cols, const float* __restrict__ nrm,
                            const float* __restrict__ dinv, float* __restrict__ AGG,
                            int e, int n) {
    int g  = blockIdx.x * blockDim.x + threadIdx.x;
    int ed = g >> 5;
    int c4 = (g & 31) * 4;
    if (ed >= e + n) return;
    int r, c; float w;
    if (ed < e) { r = rows[ed]; c = cols[ed]; w = nrm[ed]; }
    else        { r = c = ed - e; float d = dinv[r]; w = d * d; }   // self-loop
    float4 hv = *(const float4*)(H + (size_t)r * CC + c4);
    float* dst = AGG + (size_t)c * CC + c4;
    atomicAdd(dst + 0, hv.x * w);
    atomicAdd(dst + 1, hv.y * w);
    atomicAdd(dst + 2, hv.z * w);
    atomicAdd(dst + 3, hv.w * w);
}

// ---------------- BatchNorm: bias + per-channel sum / sumsq ----------------
__global__ void gcn_bn_stats(float* __restrict__ AGG, const float* __restrict__ bias,
                             float* __restrict__ sums, int n) {
    __shared__ float s0[256];
    __shared__ float s1[256];
    int c    = threadIdx.x & 127;
    int half = threadIdx.x >> 7;
    float bc = bias[c];
    float s = 0.0f, ss = 0.0f;
    for (int row = blockIdx.x * 2 + half; row < n; row += gridDim.x * 2) {
        float v = AGG[(size_t)row * CC + c] + bc;
        AGG[(size_t)row * CC + c] = v;              // keep h = agg + b for normalize pass
        s += v; ss += v * v;
    }
    s0[threadIdx.x] = s; s1[threadIdx.x] = ss;
    __syncthreads();
    if (half == 0) {
        s  += s0[threadIdx.x + 128];
        ss += s1[threadIdx.x + 128];
        atomicAdd(&sums[c], s);
        atomicAdd(&sums[128 + c], ss);
    }
}

__global__ void gcn_bn_finalize(const float* __restrict__ sums, const float* __restrict__ gam,
                                const float* __restrict__ bet, float* __restrict__ ssh,
                                float invN) {
    int c = threadIdx.x;                            // 128 threads
    float mu  = sums[c] * invN;
    float var = sums[128 + c] * invN - mu * mu;     // biased variance, matches reference
    float sc  = gam[c] * rsqrtf(var + 1e-5f);
    ssh[c]       = sc;
    ssh[128 + c] = bet[c] - mu * sc;
}

__global__ void gcn_bn_apply(const float* __restrict__ AGG, const float* __restrict__ ssh,
                             float* __restrict__ OUT, int nVec) {
    int g = blockIdx.x * blockDim.x + threadIdx.x;
    if (g >= nVec) return;
    int c4 = (g & 31) * 4;
    float4 v = ((const float4*)AGG)[g];
    v.x = fmaxf(fmaf(v.x, ssh[c4 + 0], ssh[128 + c4 + 0]), 0.0f);
    v.y = fmaxf(fmaf(v.y, ssh[c4 + 1], ssh[128 + c4 + 1]), 0.0f);
    v.z = fmaxf(fmaf(v.z, ssh[c4 + 2], ssh[128 + c4 + 2]), 0.0f);
    v.w = fmaxf(fmaf(v.w, ssh[c4 + 3], ssh[128 + c4 + 3]), 0.0f);
    ((float4*)OUT)[g] = v;
}

// ---------------- launch ----------------
extern "C" void kernel_launch(void* const* d_in, const int* in_sizes, int n_in,
                              void* d_out, int out_size, void* d_ws, size_t ws_size,
                              hipStream_t stream) {
    (void)in_sizes; (void)n_in; (void)out_size; (void)ws_size;
    const float* x   = (const float*)d_in[0];
    const int*   ei  = (const int*)d_in[1];
    const float* W   = (const float*)d_in[2];
    const float* b   = (const float*)d_in[3];
    const float* gam = (const float*)d_in[4];
    const float* bet = (const float*)d_in[5];

    char* ws = (char*)d_ws;
    size_t off = 0;
    auto alloc = [&](size_t bytes) -> void* {
        void* p = ws + off;
        off = (off + bytes + 255) & ~(size_t)255;
        return p;
    };
    float*     dinv = (float*)alloc((size_t)NN * 4);
    float*     nrm  = (float*)alloc((size_t)EE * 4);
    _Float16*  Wt   = (_Float16*)alloc((size_t)LL * CC * CC * 2);
    float*     sums = (float*)alloc(256 * 4);
    float*     ssh  = (float*)alloc(256 * 4);
    float*     bufX = (float*)alloc((size_t)NN * CC * 4);
    float*     bufH = (float*)alloc((size_t)NN * CC * 4);
    float*     bufA = (float*)alloc((size_t)NN * CC * 4);

    const int* rows = ei;           // edge_index[0] : source
    const int* cols = ei + EE;      // edge_index[1] : target (aggregation index)

    gcn_wcvt<<<(LL * CC * CC + 255) / 256, 256, 0, stream>>>(W, Wt, LL * CC * CC);
    gcn_deg_init<<<(NN + 255) / 256, 256, 0, stream>>>(dinv, NN);
    gcn_deg_count<<<(EE + 255) / 256, 256, 0, stream>>>(cols, dinv, EE);
    gcn_deg_rsqrt<<<(NN + 255) / 256, 256, 0, stream>>>(dinv, NN);
    gcn_norm<<<(EE + 255) / 256, 256, 0, stream>>>(rows, cols, dinv, nrm, EE);

    const int strips = NN / 16;     // 3125, exact
    for (int l = 0; l < LL; ++l) {
        const float* xin = (l == 0) ? x : bufX;
        gcn_gemm_wmma<<<(strips + 7) / 8, 256, 0, stream>>>(
            xin, Wt + (size_t)l * CC * CC, bufH, strips);

        hipMemsetAsync(bufA, 0, (size_t)NN * CC * 4, stream);
        hipMemsetAsync(sums, 0, 256 * 4, stream);

        int scth = (EE + NN) * 32;
        gcn_scatter<<<(scth + 255) / 256, 256, 0, stream>>>(
            bufH, rows, cols, nrm, dinv, bufA, EE, NN);

        gcn_bn_stats<<<512, 256, 0, stream>>>(bufA, b + (size_t)l * CC, sums, NN);
        gcn_bn_finalize<<<1, 128, 0, stream>>>(
            sums, gam + (size_t)l * CC, bet + (size_t)l * CC, ssh, 1.0f / NN);

        float* outp = (l == LL - 1) ? (float*)d_out : bufX;
        int nVec = NN * CC / 4;
        gcn_bn_apply<<<(nVec + 255) / 256, 256, 0, stream>>>(bufA, ssh, outp, nVec);
    }
}
